// DGN_actor_56143812493411
// MI455X (gfx1250) — compile-verified, matrix operand order checked by reading, not run on hardware
//
#include <hip/hip_runtime.h>
#include <hip/hip_bf16.h>

// Problem sizes (fixed by the reference)
#define B_    1024
#define N_    128
#define OBS_  64
#define HID_  128
#define ACT_  9

typedef __bf16 bf16_t;
typedef __attribute__((ext_vector_type(16))) __bf16 v16bf;
typedef __attribute__((ext_vector_type(8)))  float  v8f;
typedef __attribute__((ext_vector_type(4)))  unsigned int v4u;

// ---- LDS layout: bf16-element offsets (all multiples of 8 -> 16B aligned) ----
constexpr int E_X   = 0;                    // [128][64]  x (bf16)
constexpr int E_H1  = E_X  + N_ * OBS_;     // [128][128] h1
constexpr int E_Q   = E_H1 + N_ * HID_;     // q, later att
constexpr int E_K   = E_Q  + N_ * HID_;     // k, later h2 / a3
constexpr int E_VT  = E_K  + N_ * HID_;     // v transposed [d][m]
constexpr int E_WB  = E_VT + N_ * HID_;     // weight (B^T) buffer
constexpr int E_END = E_WB + HID_ * HID_;   // = 90112 bf16 elems (180224 B)
// float (dword) offsets, starting right after the bf16 region
constexpr int F_BIAS = E_END / 2;           // 128 f32
constexpr int F_W4   = F_BIAS + HID_;       // 128*9 f32
constexpr int F_B4   = F_W4 + HID_ * ACT_;  // 9 f32
constexpr int SMEM_BYTES = (F_B4 + 16) * 4; // ~185.5 KB < 320 KB/WGP

__device__ __forceinline__ unsigned short f2bf_bits(float f) {
  return (unsigned short)(__builtin_bit_cast(unsigned int, f) >> 16);
}
__device__ __forceinline__ float bf_bits2f(unsigned short h) {
  return __builtin_bit_cast(float, ((unsigned int)h) << 16);
}

// Load a 16x32 bf16 WMMA fragment (A layout; also serves B given B^T storage)
// from row-major LDS, as two 16-byte LDS loads per lane.
// ISA 7.12.2: lane (half,row): K = {0..7, 16..23} + 8*half  (contiguous chunks).
template<int LD>
__device__ __forceinline__ v16bf load_frag(int elemBase, int lane) {
  extern __shared__ v4u lds_v4[];
  const int half = lane >> 4, row = lane & 15;
  const int q = (elemBase + row * LD + 8 * half) >> 3;   // v4u index (8 bf16 each)
  const v4u a0 = lds_v4[q];        // K offsets 0..7
  const v4u a1 = lds_v4[q + 2];    // K offsets 16..23
  union { unsigned int u[8]; v16bf v; } fu;
  fu.u[0] = a0.x; fu.u[1] = a0.y; fu.u[2] = a0.z; fu.u[3] = a0.w;
  fu.u[4] = a1.x; fu.u[5] = a1.y; fu.u[6] = a1.z; fu.u[7] = a1.w;
  return fu.v;
}

// One wave computes rows [16*wave, 16*wave+16) x 128 cols of
// Out = act(A @ B + bias), everything addressed by integer LDS offsets.
// A: [128][KDIM] row-major bf16.  BT: [128][KDIM] = B^T row-major bf16.
template<int KDIM>
__device__ __forceinline__ void gemm_tile_rows(int aBase, int btBase, int biasF,
                                               bool relu, bool transpose_out,
                                               int outBase, int wave, int lane) {
  extern __shared__ unsigned short lds_u16[];
  extern __shared__ float lds_f[];
  v8f acc[8] = {};
#pragma unroll
  for (int k = 0; k < KDIM; k += 32) {
    const v16bf a = load_frag<KDIM>(aBase + wave * 16 * KDIM + k, lane);
#pragma unroll
    for (int t = 0; t < 8; ++t) {
      const v16bf bb = load_frag<KDIM>(btBase + t * 16 * KDIM + k, lane);
      acc[t] = __builtin_amdgcn_wmma_f32_16x16x32_bf16(
          false, a, false, bb, (short)0, acc[t], false, false);
    }
  }
  const int half = lane >> 4, col0 = lane & 15;
#pragma unroll
  for (int t = 0; t < 8; ++t) {
    const int col = t * 16 + col0;
    const float bias = (biasF >= 0) ? lds_f[biasF + col] : 0.0f;
#pragma unroll
    for (int r = 0; r < 8; ++r) {
      float v = acc[t][r] + bias;
      if (relu) v = fmaxf(v, 0.0f);
      const int row = wave * 16 + r + 8 * half;   // C layout: M = r + 8*half
      const int idx = transpose_out ? (outBase + col * HID_ + row)
                                    : (outBase + row * HID_ + col);
      lds_u16[idx] = f2bf_bits(v);
    }
  }
}

// Global [din][128] f32 weight -> LDS transposed [128][din] bf16 (WMMA B^T form).
__device__ __forceinline__ void load_weight128_T(const float* __restrict__ w,
                                                 const float* __restrict__ b,
                                                 int din, int btBase, int tid) {
  extern __shared__ unsigned short lds_u16[];
  extern __shared__ float lds_f[];
  const int total = din << 7;
  for (int i = tid; i < total; i += 256) {
    const int d = i >> 7, j = i & 127;
    lds_u16[btBase + j * din + d] = f2bf_bits(w[i]);
  }
  if (b)
    for (int j = tid; j < HID_; j += 256) lds_f[F_BIAS + j] = b[j];
}

__global__ __launch_bounds__(256, 1)
void dgn_actor_fused(const float* __restrict__ x, const float* __restrict__ mask,
                     const float* __restrict__ enc_w, const float* __restrict__ enc_b,
                     const float* __restrict__ wv, const float* __restrict__ bv,
                     const float* __restrict__ wk, const float* __restrict__ bk,
                     const float* __restrict__ wq, const float* __restrict__ bq,
                     const float* __restrict__ w1, const float* __restrict__ b1,
                     const float* __restrict__ w2, const float* __restrict__ b2,
                     const float* __restrict__ w3, const float* __restrict__ b3,
                     const float* __restrict__ w4, const float* __restrict__ b4,
                     float* __restrict__ actions_out, float* __restrict__ h_out) {
  extern __shared__ unsigned short lds_u16[];
  extern __shared__ float lds_f[];

  const int b    = blockIdx.x;
  const int tid  = threadIdx.x;
  const int wave = tid >> 5;
  const int lane = tid & 31;

  // ---- Encoder: h1 = relu(x @ enc_w + enc_b) ----
  for (int i = tid; i < N_ * OBS_; i += 256)
    lds_u16[E_X + i] = f2bf_bits(x[(size_t)b * N_ * OBS_ + i]);
  load_weight128_T(enc_w, enc_b, OBS_, E_WB, tid);
  __syncthreads();
  gemm_tile_rows<OBS_>(E_X, E_WB, F_BIAS, true, false, E_H1, wave, lane);
  __syncthreads();

  // ---- q / k / v projections ----
  load_weight128_T(wq, bq, HID_, E_WB, tid); __syncthreads();
  gemm_tile_rows<HID_>(E_H1, E_WB, F_BIAS, true, false, E_Q, wave, lane); __syncthreads();
  load_weight128_T(wk, bk, HID_, E_WB, tid); __syncthreads();
  gemm_tile_rows<HID_>(E_H1, E_WB, F_BIAS, true, false, E_K, wave, lane); __syncthreads();
  load_weight128_T(wv, bv, HID_, E_WB, tid); __syncthreads();
  gemm_tile_rows<HID_>(E_H1, E_WB, F_BIAS, true, true,  E_VT, wave, lane); __syncthreads();

  // ---- scores = q @ k^T ; mask ; write h ; softmax in registers ; att -> E_Q ----
  {
    v8f acc[8] = {};
#pragma unroll
    for (int k = 0; k < HID_; k += 32) {
      const v16bf a = load_frag<HID_>(E_Q + wave * 16 * HID_ + k, lane);
#pragma unroll
      for (int t = 0; t < 8; ++t) {
        const v16bf bb = load_frag<HID_>(E_K + t * 16 * HID_ + k, lane);
        acc[t] = __builtin_amdgcn_wmma_f32_16x16x32_bf16(
            false, a, false, bb, (short)0, acc[t], false, false);
      }
    }
    const int half = lane >> 4, col0 = lane & 15;
    const size_t mbase = (size_t)b * N_ * N_;
    float rmax[8], rsum[8];
#pragma unroll
    for (int r = 0; r < 8; ++r) rmax[r] = -3.0e38f;
#pragma unroll
    for (int t = 0; t < 8; ++t) {
      const int col = t * 16 + col0;
#pragma unroll
      for (int r = 0; r < 8; ++r) {
        const int row = wave * 16 + r + 8 * half;
        const float mv = __builtin_nontemporal_load(&mask[mbase + (size_t)row * N_ + col]);
        const float s  = acc[t][r];
        const float h  = fminf(fmaxf(s * mv, 0.0f), 9.0e13f) - 9.0e15f * (1.0f - mv);
        __builtin_nontemporal_store(h, &h_out[mbase + (size_t)row * N_ + col]);
        acc[t][r] = h;
        rmax[r]   = fmaxf(rmax[r], h);
      }
    }
    // each row lives in one 16-lane half: xor masks 1,2,4,8 reduce within it
#pragma unroll
    for (int r = 0; r < 8; ++r) {
#pragma unroll
      for (int m = 1; m < 16; m <<= 1)
        rmax[r] = fmaxf(rmax[r], __shfl_xor(rmax[r], m, 32));
      rsum[r] = 0.0f;
    }
#pragma unroll
    for (int t = 0; t < 8; ++t)
#pragma unroll
      for (int r = 0; r < 8; ++r) {
        const float e = __expf(acc[t][r] - rmax[r]);
        acc[t][r] = e;
        rsum[r] += e;
      }
#pragma unroll
    for (int r = 0; r < 8; ++r) {
#pragma unroll
      for (int m = 1; m < 16; m <<= 1)
        rsum[r] += __shfl_xor(rsum[r], m, 32);
      rsum[r] = 1.0f / rsum[r];
    }
    __syncthreads();   // all q/k reads complete before overwriting E_Q with att
#pragma unroll
    for (int t = 0; t < 8; ++t) {
      const int col = t * 16 + col0;
#pragma unroll
      for (int r = 0; r < 8; ++r) {
        const int row = wave * 16 + r + 8 * half;
        lds_u16[E_Q + row * HID_ + col] = f2bf_bits(acc[t][r] * rsum[r]);
      }
    }
  }
  __syncthreads();

  // ---- h2 = att @ v  (att in E_Q, v^T in E_VT; result -> E_K) ----
  gemm_tile_rows<HID_>(E_Q, E_VT, -1, false, false, E_K, wave, lane);
  __syncthreads();

  // ---- Actor MLP ----
  load_weight128_T(w1, b1, HID_, E_WB, tid); __syncthreads();
  gemm_tile_rows<HID_>(E_K, E_WB, F_BIAS, true, false, E_H1, wave, lane); __syncthreads();
  load_weight128_T(w2, b2, HID_, E_WB, tid); __syncthreads();
  gemm_tile_rows<HID_>(E_H1, E_WB, F_BIAS, true, false, E_Q, wave, lane); __syncthreads();
  load_weight128_T(w3, b3, HID_, E_WB, tid); __syncthreads();
  gemm_tile_rows<HID_>(E_Q, E_WB, F_BIAS, true, false, E_K, wave, lane);
  for (int i = tid; i < HID_ * ACT_; i += 256) lds_f[F_W4 + i] = w4[i];
  for (int i = tid; i < ACT_;        i += 256) lds_f[F_B4 + i] = b4[i];
  __syncthreads();

  // ---- logits = a3 @ w4 + b4 ; Gumbel-max categorical sample ----
  if (tid < N_) {
    float logits[ACT_];
#pragma unroll
    for (int j = 0; j < ACT_; ++j) logits[j] = lds_f[F_B4 + j];
    for (int d = 0; d < HID_; ++d) {
      const float a = bf_bits2f(lds_u16[E_K + tid * HID_ + d]);
#pragma unroll
      for (int j = 0; j < ACT_; ++j) logits[j] += a * lds_f[F_W4 + d * ACT_ + j];
    }
    const unsigned int seed = ((unsigned int)(b * N_ + tid)) * 2654435761u ^ 42u;
    int best = 0; float bestv = -3.0e38f;
#pragma unroll
    for (int j = 0; j < ACT_; ++j) {
      unsigned int h = seed + (unsigned int)j * 0x9E3779B9u;
      h ^= h >> 16; h *= 0x7feb352du; h ^= h >> 15; h *= 0x846ca68bu; h ^= h >> 16;
      float u = (float)(h >> 8) * (1.0f / 16777216.0f);
      u = fminf(fmaxf(u, 1.0e-7f), 0.999999f);
      const float g = -__logf(-__logf(u));
      const float val = logits[j] + g;
      if (val > bestv) { bestv = val; best = j; }
    }
    actions_out[(size_t)b * N_ + tid] = (float)best;
  }
}

extern "C" void kernel_launch(void* const* d_in, const int* in_sizes, int n_in,
                              void* d_out, int out_size, void* d_ws, size_t ws_size,
                              hipStream_t stream) {
  const float* x     = (const float*)d_in[0];
  const float* mask  = (const float*)d_in[1];
  const float* enc_w = (const float*)d_in[2];
  const float* enc_b = (const float*)d_in[3];
  const float* wv = (const float*)d_in[4];  const float* bv = (const float*)d_in[5];
  const float* wk = (const float*)d_in[6];  const float* bk = (const float*)d_in[7];
  const float* wq = (const float*)d_in[8];  const float* bq = (const float*)d_in[9];
  const float* w1 = (const float*)d_in[10]; const float* b1 = (const float*)d_in[11];
  const float* w2 = (const float*)d_in[12]; const float* b2 = (const float*)d_in[13];
  const float* w3 = (const float*)d_in[14]; const float* b3 = (const float*)d_in[15];
  const float* w4 = (const float*)d_in[16]; const float* b4 = (const float*)d_in[17];

  float* actions_out = (float*)d_out;                     // [B,N]   (output 0)
  float* h_out       = actions_out + (size_t)B_ * N_;     // [B,N,N] (output 1)

  dgn_actor_fused<<<dim3(B_), dim3(256), SMEM_BYTES, stream>>>(
      x, mask, enc_w, enc_b, wv, bv, wk, bk, wq, bq,
      w1, b1, w2, b2, w3, b3, w4, b4, actions_out, h_out);
}
